// ConflictGraphEncoder_80676665688154
// MI455X (gfx1250) — matrix operand order repaired and checked by reading, not compiled, not run on hardware
//
#include <hip/hip_runtime.h>

typedef __attribute__((ext_vector_type(16))) __bf16 v16bf;
typedef __attribute__((ext_vector_type(8)))  float  v8f;

union Frag16 {
    v16bf v;
    uint4 q[2];
    __bf16 h[16];
};

__device__ __forceinline__ v8f wmma_bf16(const Frag16& a, const Frag16& b, v8f c) {
    return __builtin_amdgcn_wmma_f32_16x16x32_bf16(false, a.v, false, b.v,
                                                   (short)0, c, false, false);
}

// ---------------------------------------------------------------- utilities
__global__ void zero_kernel(float* __restrict__ sums, float* __restrict__ counts,
                            long long n_sums, long long n_counts) {
    long long i = (long long)blockIdx.x * blockDim.x + threadIdx.x;
    if (i < n_sums)   sums[i]   = 0.0f;
    if (i < n_counts) counts[i] = 0.0f;
}

__global__ void count_kernel(const long long* __restrict__ eidx,
                             float* __restrict__ counts, long long E) {
    long long e = (long long)blockIdx.x * blockDim.x + threadIdx.x;
    if (e < E) atomicAdd(&counts[eidx[E + e]], 1.0f);
}

// ---------------------------------------------------------------- node MLP (tiny: 6->64->64)
__global__ __launch_bounds__(256)
void node_mlp_kernel(const float* __restrict__ x,
                     const float* __restrict__ W1, const float* __restrict__ b1,
                     const float* __restrict__ W2, const float* __restrict__ b2,
                     float* __restrict__ node_emb, __bf16* __restrict__ node_bf,
                     long long N) {
    __shared__ float h1[4][64];
    int t = threadIdx.x;
    int ln = t >> 6;
    int c  = t & 63;
    long long node = (long long)blockIdx.x * 4 + ln;
    bool act = node < N;
    if (act) {
        const float* xr = x + node * 6;
        float acc = b1[c];
        #pragma unroll
        for (int i = 0; i < 6; ++i) acc += xr[i] * W1[i * 64 + c];
        h1[ln][c] = fmaxf(acc, 0.0f);
    }
    __syncthreads();
    if (act) {
        float acc = b2[c];
        #pragma unroll
        for (int k = 0; k < 64; ++k) acc += h1[ln][k] * W2[k * 64 + c];
        long long o = node * 64 + c;
        node_emb[o] = acc;
        node_bf[o]  = (__bf16)acc;
    }
}

// ---------------------------------------------------------------- edge MLP (4->64 scalar, 64->64 WMMA)
__global__ __launch_bounds__(128)
void edge_mlp_kernel(const float* __restrict__ xe,
                     const float* __restrict__ W1, const float* __restrict__ b1,
                     const float* __restrict__ W2, const float* __restrict__ b2,
                     float* __restrict__ edge_emb, long long E) {
    __shared__ __align__(16) __bf16 Wt2[64 * 64];      // Wt2[n][k] = W2[k][n]
    __shared__ __align__(16) __bf16 hs[4][16 * 64];    // per-wave staged h1 (bf16)
    int t = threadIdx.x;
    for (int i = t; i < 64 * 64; i += 128) {
        int n = i >> 6, k = i & 63;
        Wt2[i] = (__bf16)W2[k * 64 + n];
    }
    int wave = t >> 5, lane = t & 31;
    long long ebase = ((long long)blockIdx.x * 4 + wave) * 16;
    bool act = ebase < E;
    int m = lane & 15, g = lane >> 4;
    long long e = ebase + m; if (e >= E) e = E - 1;
    if (act) {
        const float* xr = xe + e * 4;
        float x0 = xr[0], x1 = xr[1], x2 = xr[2], x3 = xr[3];
        #pragma unroll
        for (int kk = 0; kk < 32; ++kk) {
            int k = g * 32 + kk;
            float h = b1[k] + x0 * W1[k] + x1 * W1[64 + k] + x2 * W1[128 + k] + x3 * W1[192 + k];
            hs[wave][m * 64 + k] = (__bf16)fmaxf(h, 0.0f);
        }
    }
    __syncthreads();
    if (!act) return;

    v8f acc[4];
    #pragma unroll
    for (int nt = 0; nt < 4; ++nt)
        #pragma unroll
        for (int r = 0; r < 8; ++r) acc[nt][r] = 0.0f;

    const __bf16* rp = &hs[wave][m * 64];
    #pragma unroll
    for (int ks = 0; ks < 2; ++ks) {
        Frag16 a;
        int kb = ks * 32 + g * 8;
        a.q[0] = *(const uint4*)(rp + kb);
        a.q[1] = *(const uint4*)(rp + kb + 16);
        #pragma unroll
        for (int nt = 0; nt < 4; ++nt) {
            Frag16 bf;
            const __bf16* wp = &Wt2[(nt * 16 + m) * 64 + ks * 32 + g * 8];
            bf.q[0] = *(const uint4*)wp;
            bf.q[1] = *(const uint4*)(wp + 16);
            acc[nt] = wmma_bf16(a, bf, acc[nt]);
        }
    }
    int n = lane & 15;
    #pragma unroll
    for (int r = 0; r < 8; ++r) {
        long long e2 = ebase + r + 8 * g;
        if (e2 >= E) continue;
        float* er = edge_emb + e2 * 64;
        #pragma unroll
        for (int nt = 0; nt < 4; ++nt) {
            int c = nt * 16 + n;
            er[c] = acc[nt][r] + b2[c];
        }
    }
}

// ---------------------------------------------------------------- GNN layer edge kernel ([E,128]@[128,64] WMMA + scatter)
__global__ __launch_bounds__(128)
void gnn_edge_kernel(const __bf16* __restrict__ node_bf,
                     const long long* __restrict__ eidx,
                     float* __restrict__ edge_emb,
                     float* __restrict__ sums,
                     const float* __restrict__ W, const float* __restrict__ b,
                     long long E) {
    __shared__ __align__(16) __bf16 Wt[64 * 128];   // Wt[n][k] = W[k][n], K=128
    int t = threadIdx.x;
    for (int i = t; i < 64 * 128; i += 128) {
        int n = i >> 7, k = i & 127;
        Wt[i] = (__bf16)W[k * 64 + n];
    }
    __syncthreads();
    int wave = t >> 5, lane = t & 31;
    long long ebase = ((long long)blockIdx.x * 4 + wave) * 16;
    if (ebase >= E) return;
    int m = lane & 15, g = lane >> 4;
    long long e = ebase + m; if (e >= E) e = E - 1;
    const __bf16* srow = node_bf + eidx[e] * 64;       // K 0..63
    const __bf16* drow = node_bf + eidx[E + e] * 64;   // K 64..127

    v8f acc[4];
    #pragma unroll
    for (int nt = 0; nt < 4; ++nt)
        #pragma unroll
        for (int r = 0; r < 8; ++r) acc[nt][r] = 0.0f;

    #pragma unroll
    for (int ks = 0; ks < 4; ++ks) {
        const __bf16* rp = (ks < 2) ? srow : drow;
        int kb = (ks & 1) * 32 + g * 8;
        Frag16 a;
        a.q[0] = *(const uint4*)(rp + kb);
        a.q[1] = *(const uint4*)(rp + kb + 16);
        #pragma unroll
        for (int nt = 0; nt < 4; ++nt) {
            Frag16 bf;
            const __bf16* wp = &Wt[(nt * 16 + m) * 128 + ks * 32 + g * 8];
            bf.q[0] = *(const uint4*)wp;
            bf.q[1] = *(const uint4*)(wp + 16);
            acc[nt] = wmma_bf16(a, bf, acc[nt]);
        }
    }
    int n = lane & 15;
    #pragma unroll
    for (int r = 0; r < 8; ++r) {
        long long e2 = ebase + r + 8 * g;
        if (e2 >= E) continue;
        long long d2 = eidx[E + e2];
        float* er = edge_emb + e2 * 64;
        float* sr = sums + d2 * 64;
        #pragma unroll
        for (int nt = 0; nt < 4; ++nt) {
            int c = nt * 16 + n;
            float msg = fmaxf(acc[nt][r] + b[c], 0.0f);
            float v = er[c] + msg;
            er[c] = v;
            atomicAdd(&sr[c], v);
        }
    }
}

// ---------------------------------------------------------------- node update (mean-aggregate + residual + re-zero)
__global__ void node_update_kernel(float* __restrict__ node_emb, __bf16* __restrict__ node_bf,
                                   float* __restrict__ sums, const float* __restrict__ counts,
                                   long long NH) {
    long long i = (long long)blockIdx.x * blockDim.x + threadIdx.x;
    if (i >= NH) return;
    long long node = i >> 6;
    float c = counts[node];
    float v = node_emb[i] + (c > 0.0f ? sums[i] / c : 0.0f);
    node_emb[i] = v;
    node_bf[i]  = (__bf16)v;
    sums[i] = 0.0f;
}

// ---------------------------------------------------------------- output MLP (64->64 WMMA, 64->3 scalar)
__global__ __launch_bounds__(128)
void out_mlp_kernel(const float* __restrict__ edge_emb,
                    const float* __restrict__ W1, const float* __restrict__ b1,
                    const float* __restrict__ W2, const float* __restrict__ b2,
                    float* __restrict__ out, long long E) {
    __shared__ __align__(16) __bf16 Wt1[64 * 64];
    __shared__ float hsf[4][16 * 64];
    __shared__ float W2l[64 * 3];
    int t = threadIdx.x;
    for (int i = t; i < 64 * 64; i += 128) {
        int n = i >> 6, k = i & 63;
        Wt1[i] = (__bf16)W1[k * 64 + n];
    }
    for (int i = t; i < 192; i += 128) W2l[i] = W2[i];
    __syncthreads();
    int wave = t >> 5, lane = t & 31;
    long long ebase = ((long long)blockIdx.x * 4 + wave) * 16;
    bool act = ebase < E;
    int m = lane & 15, g = lane >> 4;
    long long e = ebase + m; if (e >= E) e = E - 1;
    if (act) {
        const float* row = edge_emb + e * 64;
        v8f acc[4];
        #pragma unroll
        for (int nt = 0; nt < 4; ++nt)
            #pragma unroll
            for (int r = 0; r < 8; ++r) acc[nt][r] = 0.0f;
        #pragma unroll
        for (int ks = 0; ks < 2; ++ks) {
            Frag16 a;
            int kb = ks * 32 + g * 8;
            #pragma unroll
            for (int j = 0; j < 8; ++j) a.h[j]     = (__bf16)row[kb + j];
            #pragma unroll
            for (int j = 0; j < 8; ++j) a.h[8 + j] = (__bf16)row[kb + 16 + j];
            #pragma unroll
            for (int nt = 0; nt < 4; ++nt) {
                Frag16 bf;
                const __bf16* wp = &Wt1[(nt * 16 + m) * 64 + ks * 32 + g * 8];
                bf.q[0] = *(const uint4*)wp;
                bf.q[1] = *(const uint4*)(wp + 16);
                acc[nt] = wmma_bf16(a, bf, acc[nt]);
            }
        }
        int n = lane & 15;
        #pragma unroll
        for (int r = 0; r < 8; ++r) {
            int m2 = r + 8 * g;
            #pragma unroll
            for (int nt = 0; nt < 4; ++nt) {
                int c = nt * 16 + n;
                hsf[wave][m2 * 64 + c] = fmaxf(acc[nt][r] + b1[c], 0.0f);
            }
        }
    }
    __syncthreads();
    if (act && lane < 16) {
        long long e3 = ebase + lane;
        if (e3 < E) {
            float o0 = b2[0], o1 = b2[1], o2 = b2[2];
            const float* hr = &hsf[wave][lane * 64];
            #pragma unroll
            for (int k = 0; k < 64; ++k) {
                float h = hr[k];
                o0 += h * W2l[k * 3 + 0];
                o1 += h * W2l[k * 3 + 1];
                o2 += h * W2l[k * 3 + 2];
            }
            out[e3 * 3 + 0] = o0;
            out[e3 * 3 + 1] = o1;
            out[e3 * 3 + 2] = o2;
        }
    }
}

// ---------------------------------------------------------------- host
static inline size_t alignUp256(size_t x) { return (x + 255) & ~(size_t)255; }

extern "C" void kernel_launch(void* const* d_in, const int* in_sizes, int n_in,
                              void* d_out, int out_size, void* d_ws, size_t ws_size,
                              hipStream_t stream) {
    // Detect flattening order: insertion order puts edge_indices (2E) before
    // edge_features (4E); tree_flatten-sorted puts edge_features first.
    bool sortedOrder = in_sizes[1] > in_sizes[2];
    int iA = 0, iEI, iEF, iNW1, iNB1, iNW2, iNB2, iEW1, iEB1, iEW2, iEB2;
    int iGW[3], iGB[3], iOW1, iOB1, iOW2, iOB2;
    if (!sortedOrder) {
        iEI = 1; iEF = 2;
        iNW1 = 3; iNB1 = 4; iNW2 = 5; iNB2 = 6;
        iEW1 = 7; iEB1 = 8; iEW2 = 9; iEB2 = 10;
        iGW[0] = 11; iGW[1] = 12; iGW[2] = 13;
        iGB[0] = 14; iGB[1] = 15; iGB[2] = 16;
        iOW1 = 17; iOB1 = 18; iOW2 = 19; iOB2 = 20;
    } else {
        iEF = 1; iEI = 2;
        iEW1 = 3; iEW2 = 4; iEB1 = 5; iEB2 = 6;
        iGW[0] = 7; iGW[1] = 8; iGW[2] = 9;
        iGB[0] = 10; iGB[1] = 11; iGB[2] = 12;
        iNW1 = 13; iNW2 = 14; iNB1 = 15; iNB2 = 16;
        iOW1 = 17; iOW2 = 18; iOB1 = 19; iOB2 = 20;
    }
    long long N = in_sizes[iA] / 6;
    long long E = in_sizes[iEF] / 4;

    const float*     agent = (const float*)d_in[iA];
    const long long* eidx  = (const long long*)d_in[iEI];
    const float*     efeat = (const float*)d_in[iEF];
    const float* nW1 = (const float*)d_in[iNW1]; const float* nb1 = (const float*)d_in[iNB1];
    const float* nW2 = (const float*)d_in[iNW2]; const float* nb2 = (const float*)d_in[iNB2];
    const float* eW1 = (const float*)d_in[iEW1]; const float* eb1 = (const float*)d_in[iEB1];
    const float* eW2 = (const float*)d_in[iEW2]; const float* eb2 = (const float*)d_in[iEB2];
    const float* oW1 = (const float*)d_in[iOW1]; const float* ob1 = (const float*)d_in[iOB1];
    const float* oW2 = (const float*)d_in[iOW2]; const float* ob2 = (const float*)d_in[iOB2];

    char* base = (char*)d_ws;
    size_t off = 0;
    float*  node_emb = (float*)(base + off);  off += alignUp256((size_t)N * 64 * sizeof(float));
    __bf16* node_bf  = (__bf16*)(base + off); off += alignUp256((size_t)N * 64 * sizeof(__bf16));
    float*  sums     = (float*)(base + off);  off += alignUp256((size_t)N * 64 * sizeof(float));
    float*  counts   = (float*)(base + off);  off += alignUp256((size_t)N * sizeof(float));
    float*  edge_emb = (float*)(base + off);  off += alignUp256((size_t)E * 64 * sizeof(float));
    (void)ws_size; (void)n_in; (void)out_size;

    long long NH = N * 64;

    zero_kernel<<<(unsigned)((NH + 255) / 256), 256, 0, stream>>>(sums, counts, NH, N);
    count_kernel<<<(unsigned)((E + 255) / 256), 256, 0, stream>>>(eidx, counts, E);
    node_mlp_kernel<<<(unsigned)((N + 3) / 4), 256, 0, stream>>>(agent, nW1, nb1, nW2, nb2,
                                                                 node_emb, node_bf, N);
    long long tiles  = (E + 15) / 16;
    long long blocks = (tiles + 3) / 4;
    edge_mlp_kernel<<<(unsigned)blocks, 128, 0, stream>>>(efeat, eW1, eb1, eW2, eb2, edge_emb, E);

    for (int l = 0; l < 3; ++l) {
        gnn_edge_kernel<<<(unsigned)blocks, 128, 0, stream>>>(node_bf, eidx, edge_emb, sums,
                                                              (const float*)d_in[iGW[l]],
                                                              (const float*)d_in[iGB[l]], E);
        node_update_kernel<<<(unsigned)((NH + 255) / 256), 256, 0, stream>>>(node_emb, node_bf,
                                                                             sums, counts, NH);
    }
    out_mlp_kernel<<<(unsigned)blocks, 128, 0, stream>>>(edge_emb, oW1, ob1, oW2, ob2,
                                                         (float*)d_out, E);
}
